// AsymConvLayer_81535659147297
// MI455X (gfx1250) — matrix operand order repaired and checked by reading, not compiled, not run on hardware
//
#include <hip/hip_runtime.h>
#include <math.h>

#define NROWS   200000
#define KTAPS   9
#define HALF_KK 288          // 576/2 contraction elements per LDS half
#define WPAD    296          // 288 + 8 halves pad -> 592B/col, 16B aligned, conflict-free
#define EPSV    1e-5f
#define SLOPE   0.01f

typedef __attribute__((ext_vector_type(16))) __bf16 v16bf;
typedef __attribute__((ext_vector_type(8)))  __bf16 v8bf;
typedef __attribute__((ext_vector_type(8)))  float  v8f;

// ---------------------------------------------------------------------------
// Gather + GEMM via WMMA bf16 (f32 accum).
//   out[n,d] = cbias[d] + sum_k sum_c (feat[idx[n,k],c] * in_scale[c]) * w[k,c,d]
// then LeakyReLU, store pre-BN activation, accumulate per-channel BN stats.
// in_scale folds the *previous* layer's BN scale into the weights (staged once
// per block into LDS); cbias folds the previous BN bias (constant per channel).
// ---------------------------------------------------------------------------
__global__ __launch_bounds__(256) void conv_gather_wmma(
    const float* __restrict__ feat, const int* __restrict__ idx,
    const float* __restrict__ w, const float* __restrict__ in_scale,
    const float* __restrict__ cbias, float* __restrict__ outT,
    float* __restrict__ gsum, float* __restrict__ gsq)
{
    __shared__ __bf16 lds_w[64 * WPAD];
    __shared__ float  lds_sum[64];
    __shared__ float  lds_sq[64];

    const int  tid  = threadIdx.x;
    const int  lane = tid & 31;
    const int  wv   = tid >> 5;
    const int  m    = lane & 15;          // row/col within 16x16 tile
    const bool hi   = lane >= 16;
    const int  rowbase = blockIdx.x * 128 + wv * 16;
    const int  rowg = rowbase + m;
    const int  rowc = rowg < NROWS ? rowg : NROWS - 1;   // clamp for tail gathers

    if (tid < 64) { lds_sum[tid] = 0.f; lds_sq[tid] = 0.f; }

    v8f acc[4] = {v8f{}, v8f{}, v8f{}, v8f{}};

    for (int h = 0; h < 2; ++h) {
        __syncthreads();
        // Stage weight half h as bf16 (scaled by folded BN scale), transposed
        // [cout][kkL]+pad. float4 per thread -> coalesced global_load_b128.
        for (int i = tid; i < HALF_KK * 16; i += 256) {
            const int co4 = i & 15;              // group of 4 cout
            const int kkL = i >> 4;
            const int kk  = h * HALF_KK + kkL;
            const float s = in_scale[kk & 63];
            float4 wq = ((const float4*)(w + (size_t)kk * 64))[co4];
            const int c = co4 * 4;
            lds_w[(c + 0) * WPAD + kkL] = (__bf16)(wq.x * s);
            lds_w[(c + 1) * WPAD + kkL] = (__bf16)(wq.y * s);
            lds_w[(c + 2) * WPAD + kkL] = (__bf16)(wq.z * s);
            lds_w[(c + 3) * WPAD + kkL] = (__bf16)(wq.w * s);
        }
        __syncthreads();

        for (int sL = 0; sL < 9; ++sL) {
            const int s   = h * 9 + sL;       // global 32-wide k-step, 0..17
            const int k   = s >> 1;           // conv tap 0..8
            const int sh  = s & 1;            // which 32-channel half of Cin
            const int src = idx[(size_t)rowc * KTAPS + k];

            // A fragment: 16-bit A 16x32 layout -> two contiguous 8-float runs.
            const int c0 = sh * 32 + (hi ? 8 : 0);
            const float* base = feat + (size_t)src * 64;
            const float4* p = (const float4*)(base + c0);
            const float4* q = (const float4*)(base + c0 + 16);
            float4 f0 = p[0], f1 = p[1];
            float4 g0 = q[0], g1 = q[1];
            v16bf a;
            a[0]=(__bf16)f0.x;  a[1]=(__bf16)f0.y;  a[2]=(__bf16)f0.z;  a[3]=(__bf16)f0.w;
            a[4]=(__bf16)f1.x;  a[5]=(__bf16)f1.y;  a[6]=(__bf16)f1.z;  a[7]=(__bf16)f1.w;
            a[8]=(__bf16)g0.x;  a[9]=(__bf16)g0.y;  a[10]=(__bf16)g0.z; a[11]=(__bf16)g0.w;
            a[12]=(__bf16)g1.x; a[13]=(__bf16)g1.y; a[14]=(__bf16)g1.z; a[15]=(__bf16)g1.w;

            const int kkB = sL * 32 + (hi ? 8 : 0);    // B 32x16 layout
            #pragma unroll
            for (int t = 0; t < 4; ++t) {
                const v8bf* bp = (const v8bf*)&lds_w[(t * 16 + m) * WPAD + kkB];
                v8bf blo = bp[0], bhi = bp[2];
                v16bf b = __builtin_shufflevector(blo, bhi,
                            0,1,2,3,4,5,6,7,8,9,10,11,12,13,14,15);
                acc[t] = __builtin_amdgcn_wmma_f32_16x16x32_bf16(
                            false, a, false, b, (short)0, acc[t], false, false);
            }
        }
    }

    // Epilogue: +folded bias, LeakyReLU, store, per-channel partial stats.
    #pragma unroll
    for (int t = 0; t < 4; ++t) {
        const int col = t * 16 + m;
        const float cb = cbias[col];
        float lsum = 0.f, lsq = 0.f;
        #pragma unroll
        for (int r = 0; r < 8; ++r) {
            const int row = rowbase + r + (hi ? 8 : 0);
            float v = acc[t][r] + cb;
            v = v >= 0.f ? v : SLOPE * v;
            if (row < NROWS) {
                outT[(size_t)row * 64 + col] = v;
                lsum += v;
                lsq  += v * v;
            }
        }
        atomicAdd(&lds_sum[col], lsum);
        atomicAdd(&lds_sq[col],  lsq);
    }
    __syncthreads();
    if (tid < 64)        atomicAdd(&gsum[tid], lds_sum[tid]);
    else if (tid < 128)  atomicAdd(&gsq[tid - 64], lds_sq[tid - 64]);
}

// stats -> {scale, bias}; optionally fold bias through next layer's weights:
// cbias_out[d] = sum_kk w_next[kk,d] * bias[kk % 64]
__global__ void bn_finalize_fold(const float* __restrict__ gsum, const float* __restrict__ gsq,
                                 const float* __restrict__ gamma, const float* __restrict__ beta,
                                 float* __restrict__ sb,
                                 const float* __restrict__ w_next, float* __restrict__ cbias_out)
{
    __shared__ float sbias[64];
    const int c = threadIdx.x;                 // 64 threads
    const float inv_n = 1.0f / (float)NROWS;
    const float mean  = gsum[c] * inv_n;
    const float var   = gsq[c] * inv_n - mean * mean;
    const float sc    = gamma[c] * rsqrtf(var + EPSV);
    const float bi    = beta[c] - mean * sc;
    sb[c]      = sc;
    sb[64 + c] = bi;
    sbias[c]   = bi;
    if (w_next != nullptr) {                   // uniform branch
        __syncthreads();
        float acc = 0.f;
        for (int kk = 0; kk < 576; ++kk)       // threads read contiguous 256B/step
            acc += w_next[(size_t)kk * 64 + c] * sbias[kk & 63];
        cbias_out[c] = acc;
    }
}

// o[i] = t[i]*scale + bias   (float4 per thread; channel period 64)
__global__ __launch_bounds__(256) void bn_apply(const float* __restrict__ t,
                                                const float* __restrict__ sb,
                                                float* __restrict__ o)
{
    const int i  = blockIdx.x * 256 + threadIdx.x;
    const int c0 = (i * 4) & 63;
    float4 v = ((const float4*)t)[i];
    v.x = v.x * sb[c0]     + sb[64 + c0];
    v.y = v.y * sb[c0 + 1] + sb[64 + c0 + 1];
    v.z = v.z * sb[c0 + 2] + sb[64 + c0 + 2];
    v.w = v.w * sb[c0 + 3] + sb[64 + c0 + 3];
    ((float4*)o)[i] = v;
}

// io[i] += t2[i]*scale + bias   -- final out = A + bn(lrelu(convB1)), in place
__global__ __launch_bounds__(256) void bn_apply_add(const float* __restrict__ t2,
                                                    const float* __restrict__ sb,
                                                    float* __restrict__ io)
{
    const int i  = blockIdx.x * 256 + threadIdx.x;
    const int c0 = (i * 4) & 63;
    float4 v = ((const float4*)t2)[i];
    float4 a = ((const float4*)io)[i];
    a.x += v.x * sb[c0]     + sb[64 + c0];
    a.y += v.y * sb[c0 + 1] + sb[64 + c0 + 1];
    a.z += v.z * sb[c0 + 2] + sb[64 + c0 + 2];
    a.w += v.w * sb[c0 + 3] + sb[64 + c0 + 3];
    ((float4*)io)[i] = a;
}

// Re-initialize stats region each launch: [0,384)=0 (sums), [384,448)=1 (ident
// scale), [448,512)=0 (zero cbias). Deterministic across graph replays.
__global__ void init_stats(float* __restrict__ p)
{
    const int i = blockIdx.x * blockDim.x + threadIdx.x;
    if (i < 512) p[i] = (i >= 384 && i < 448) ? 1.f : 0.f;
}

extern "C" void kernel_launch(void* const* d_in, const int* in_sizes, int n_in,
                              void* d_out, int out_size, void* d_ws, size_t ws_size,
                              hipStream_t stream)
{
    const float* x        = (const float*)d_in[0];
    const float* w_a0     = (const float*)d_in[1];
    const float* gamma_a0 = (const float*)d_in[2];
    const float* beta_a0  = (const float*)d_in[3];
    const float* w_a1     = (const float*)d_in[4];
    const float* gamma_a1 = (const float*)d_in[5];
    const float* beta_a1  = (const float*)d_in[6];
    // d_in[7..9] = w_b0/gamma_b0/beta_b0 : dead branch (DCE'd in reference)
    const float* w_b1     = (const float*)d_in[10];
    const float* gamma_b1 = (const float*)d_in[11];
    const float* beta_b1  = (const float*)d_in[12];
    const int*   idx133   = (const int*)d_in[13];
    const int*   idx313   = (const int*)d_in[14];
    float* out = (float*)d_out;

    char*  ws    = (char*)d_ws;
    float* bufA  = (float*)ws;                                     // [N,64] f32
    float* stats = (float*)(ws + (size_t)NROWS * 64 * sizeof(float));
    float* sum0 = stats,       *sq0 = stats + 64;
    float* sum1 = stats + 128, *sq1 = stats + 192;
    float* sum2 = stats + 256, *sq2 = stats + 320;
    float* ident  = stats + 384;   // scale = 1 (layers with unfolded input)
    float* zeros  = stats + 448;   // cbias = 0
    float* sb0    = stats + 512;   // {scale[64], bias[64]}
    float* sb1    = stats + 640;
    float* sb2    = stats + 768;
    float* cbias1 = stats + 896;   // folded bias constant for layer a1

    const int convGrid = (NROWS + 127) / 128;       // 1563
    const int ewGrid   = (NROWS * 64 / 4) / 256;    // 12500

    init_stats<<<2, 256, 0, stream>>>(stats);

    // Layer A0: t0 = lrelu(conv(x, idx_133, w_a0)) -> d_out ; stats0
    conv_gather_wmma<<<convGrid, 256, 0, stream>>>(x, idx133, w_a0, ident, zeros,
                                                   out, sum0, sq0);
    // sb0 + fold bn_a0's bias through w_a1 -> cbias1
    bn_finalize_fold<<<1, 64, 0, stream>>>(sum0, sq0, gamma_a0, beta_a0, sb0,
                                           w_a1, cbias1);

    // Layer A1 with BN folded into weights: gathers raw t0, no bn_apply pass.
    // t1 = lrelu(conv(bn(t0), idx_313, w_a1)) -> bufA ; stats1
    conv_gather_wmma<<<convGrid, 256, 0, stream>>>(out, idx313, w_a1, sb0, cbias1,
                                                   bufA, sum1, sq1);
    bn_finalize_fold<<<1, 64, 0, stream>>>(sum1, sq1, gamma_a1, beta_a1, sb1,
                                           nullptr, nullptr);
    // A = bn(t1) must be materialized (used in final A + B) -> d_out
    bn_apply<<<ewGrid, 256, 0, stream>>>(bufA, sb1, out);

    // Layer B1: t2 = lrelu(conv(A, idx_133, w_b1)) -> bufA ; stats2
    conv_gather_wmma<<<convGrid, 256, 0, stream>>>(out, idx133, w_b1, ident, zeros,
                                                   bufA, sum2, sq2);
    bn_finalize_fold<<<1, 64, 0, stream>>>(sum2, sq2, gamma_b1, beta_b1, sb2,
                                           nullptr, nullptr);

    // out = A + bn(t2)   (in place on d_out)
    bn_apply_add<<<ewGrid, 256, 0, stream>>>(bufA, sb2, out);
}
